// NeuralDMD_23416161698146
// MI455X (gfx1250) — matrix operand order repaired and checked by reading, not compiled, not run on hardware
//
#include <hip/hip_runtime.h>
#include <hip/hip_bf16.h>
#include <cstdint>

// ---------------------------------------------------------------------------
// Types for CDNA5 WMMA / TDM
// ---------------------------------------------------------------------------
typedef __attribute__((ext_vector_type(16))) __bf16 v16bf;
typedef __attribute__((ext_vector_type(8)))  float  v8f;
typedef __attribute__((ext_vector_type(2)))  float  v2f;
typedef __attribute__((ext_vector_type(4)))  int    v4i;
typedef __attribute__((ext_vector_type(4)))  unsigned int u32x4;
typedef __attribute__((ext_vector_type(8)))  unsigned int u32x8;

union ABFrag { v16bf v; uint4 q[2]; };

#ifndef __has_builtin
#define __has_builtin(x) 0
#endif
#if __has_builtin(__builtin_amdgcn_global_load_async_to_lds_b128)
#define HAVE_ASYNC_COPY 1
#else
#define HAVE_ASYNC_COPY 0
#endif
#if __has_builtin(__builtin_amdgcn_s_wait_asynccnt)
#define WAIT_ASYNCCNT() __builtin_amdgcn_s_wait_asynccnt(0)
#else
#define WAIT_ASYNCCNT() asm volatile("s_wait_asynccnt 0x0" ::: "memory")
#endif

__device__ __forceinline__ uint16_t f2bf(float f) {
    uint32_t x = __float_as_uint(f);
    uint32_t r = ((x >> 16) & 1u) + 0x7FFFu;   // round-to-nearest-even
    x += r;
    return (uint16_t)(x >> 16);
}

// ---------------------------------------------------------------------------
// Tensor Data Mover: 2-D tile load global->LDS.
// D# group0/group1 per CDNA5 ISA:
//   g0: [1:0]=count=1 | [63:32]=lds_addr | [120:64]=global_addr | [127:126]=2
//   g1: data_size=2B, tensor_dim0=ld, tile_dim0=tile_k, tile_dim1=tile_m,
//       tensor_dim0_stride=ld, 2-D (groups 2/3 NULL).
// Issued by one wave; tracked by TENSORcnt.
// ---------------------------------------------------------------------------
__device__ __forceinline__ void tdm_load_2d_bf16(unsigned lds_addr,
                                                 const void* gaddr,
                                                 unsigned tile_k, unsigned tile_m,
                                                 unsigned ld) {
    unsigned long long ga = (unsigned long long)gaddr;
    u32x4 g0;
    u32x8 g1;
    g0[0] = 1u;                                                // count=1
    g0[1] = lds_addr;                                          // LDS byte addr
    g0[2] = (unsigned)ga;                                      // gaddr[31:0]
    g0[3] = (unsigned)((ga >> 32) & 0x01FFFFFFu) | 0x80000000u; // [56:32]|type=2
    g1[0] = 1u << 16;                  // workgroup_mask=0, data_size=1 (2B)
    g1[1] = (ld & 0xFFFFu) << 16;      // tensor_dim0[15:0] @ bits 63:48
    g1[2] = (ld >> 16) | (0xFFFFu << 16); // dim0[31:16] | tensor_dim1[15:0]
    g1[3] = (tile_k << 16);            // tensor_dim1[31:16]=0 | tile_dim0
    g1[4] = tile_m;                    // tile_dim1 | tile_dim2=0
    g1[5] = ld;                        // tensor_dim0_stride[31:0]
    g1[6] = 0u;                        // stride[47:32] | dim1_stride[15:0]
    g1[7] = 0u;                        // dim1_stride[47:16]
    asm volatile("tensor_load_to_lds %0, %1" :: "s"(g0), "s"(g1) : "memory");
}

// ---------------------------------------------------------------------------
// fp32 -> bf16 conversion
// ---------------------------------------------------------------------------
__global__ __launch_bounds__(256)
void k_f2bf(const float* __restrict__ in, uint16_t* __restrict__ out, long n) {
    long i = (long)blockIdx.x * 256 + threadIdx.x;
    if (i < n) out[i] = f2bf(in[i]);
}

// ---------------------------------------------------------------------------
// Tiled transpose, fp32 in -> bf16 out.  in: R x C  ->  out: C x R
// Global->LDS staging uses async-to-LDS copies when the builtin exists.
// ---------------------------------------------------------------------------
__global__ __launch_bounds__(256)
void k_transpose_f2bf(const float* __restrict__ in, uint16_t* __restrict__ out,
                      int R, int C) {
    __shared__ __align__(16) float t[32][36];   // 144B row stride, 16B aligned
    int tid = threadIdx.x;
    int bx = blockIdx.x * 32;   // input col base
    int by = blockIdx.y * 32;   // input row base
    int row = tid >> 3, seg = tid & 7;          // 32 rows x 8 segs of 4 floats
#if HAVE_ASYNC_COPY
    __builtin_amdgcn_global_load_async_to_lds_b128(
        (__attribute__((address_space(1))) v4i*)
            (in + (size_t)(by + row) * C + bx + seg * 4),
        (__attribute__((address_space(3))) v4i*)&t[row][seg * 4], 0, 0);
    WAIT_ASYNCCNT();
#else
    *(float4*)&t[row][seg * 4] =
        *(const float4*)(in + (size_t)(by + row) * C + bx + seg * 4);
#endif
    __syncthreads();
    int oc = tid >> 3;          // output row = input col
    int rs = tid & 7;           // 4 input rows each
#pragma unroll
    for (int k = 0; k < 4; ++k) {
        int ir = rs * 4 + k;
        out[(size_t)(bx + oc) * R + by + ir] = f2bf(t[ir][oc]);
    }
}

// ---------------------------------------------------------------------------
// bf16 WMMA GEMM: Cf32[M,N] (+opt bf16 mirror) = act(A[M,K]@B[K,N] + bias)
// M%64==0, N%64==0, K%32==0. 128 threads (4 waves), 64x64 tile/block,
// 32x32 per wave. Double-buffered LDS; A tile streamed by the TDM
// (tensor_load_to_lds, TENSORcnt), B tile transposed via ds_store_b16.
// ---------------------------------------------------------------------------
__global__ __launch_bounds__(128)
void k_gemm_bf16(const uint16_t* __restrict__ A, int lda,
                 const uint16_t* __restrict__ B, int ldb,
                 const float* __restrict__ bias,
                 float* __restrict__ Cf, uint16_t* __restrict__ Cb, int ldc,
                 int M, int N, int K, int leaky) {
    __shared__ __align__(16) uint16_t As[2][64 * 32];   // [m][k]
    __shared__ __align__(16) uint16_t Bs[2][64 * 32];   // transposed [n][k]
    (void)M; (void)N;
    const int tid  = threadIdx.x;
    const int lane = tid & 31;
    const int wid  = tid >> 5;
    const int m0   = blockIdx.y * 64;
    const int n0   = blockIdx.x * 64;
    const int wm   = (wid & 1) * 32;
    const int wn   = (wid >> 1) * 32;
    const int hi   = (lane >> 4) & 1;
    const int lr   = lane & 15;
    const int ksteps = K / 32;

    v8f acc[2][2] = {};

    auto stageA = [&](int kt, int buf) {
        if (tid < 32)   // wave 0 drives the DMA engine
            tdm_load_2d_bf16((unsigned)(uintptr_t)(void*)&As[buf][0],
                             A + (size_t)m0 * lda + kt * 32,
                             32u, 64u, (unsigned)lda);
    };
    auto stageB = [&](int kt, int buf) {
        const int kbase = kt * 32;
        uint16_t* bs = &Bs[buf][0];
#pragma unroll
        for (int it = 0; it < 2; ++it) {
            int idx = tid + it * 128;
            int kr = idx >> 3, sg = idx & 7;
            uint4 d = *(const uint4*)(B + (size_t)(kbase + kr) * ldb + n0 + sg * 8);
            int nb = sg * 8;
            bs[(nb + 0) * 32 + kr] = (uint16_t)(d.x & 0xffffu);
            bs[(nb + 1) * 32 + kr] = (uint16_t)(d.x >> 16);
            bs[(nb + 2) * 32 + kr] = (uint16_t)(d.y & 0xffffu);
            bs[(nb + 3) * 32 + kr] = (uint16_t)(d.y >> 16);
            bs[(nb + 4) * 32 + kr] = (uint16_t)(d.z & 0xffffu);
            bs[(nb + 5) * 32 + kr] = (uint16_t)(d.z >> 16);
            bs[(nb + 6) * 32 + kr] = (uint16_t)(d.w & 0xffffu);
            bs[(nb + 7) * 32 + kr] = (uint16_t)(d.w >> 16);
        }
    };

    stageA(0, 0);
    stageB(0, 0);
    for (int kt = 0; kt < ksteps; ++kt) {
        const int buf = kt & 1;
        if (tid < 32) __builtin_amdgcn_s_wait_tensorcnt(0);
        __syncthreads();                       // tile `buf` ready for all waves
        if (kt + 1 < ksteps) {                 // prefetch next tile into buf^1
            stageA(kt + 1, buf ^ 1);
            stageB(kt + 1, buf ^ 1);
            __builtin_prefetch(B + (size_t)(kt * 32 + 64) * ldb + n0, 0, 0);
        }
        ABFrag aF[2], bF[2];
#pragma unroll
        for (int s = 0; s < 2; ++s) {
            // A frag: lane<16 -> K 0..7 & 16..23 ; lane>=16 -> K 8..15 & 24..31
            const uint4* ra = (const uint4*)(&As[buf][0] + (wm + s * 16 + lr) * 32);
            aF[s].q[0] = ra[hi];
            aF[s].q[1] = ra[hi + 2];
            // B frag: lane<16 -> K 0..7 & 16..23 ; lane>=16 -> K 8..15 & 24..31
            const uint4* rb = (const uint4*)(&Bs[buf][0] + (wn + s * 16 + lr) * 32);
            bF[s].q[0] = rb[hi];
            bF[s].q[1] = rb[hi + 2];
        }
#pragma unroll
        for (int i = 0; i < 2; ++i)
#pragma unroll
            for (int j = 0; j < 2; ++j)
                acc[i][j] = __builtin_amdgcn_wmma_f32_16x16x32_bf16(
                    false, aF[i].v, false, bF[j].v, (short)0, acc[i][j],
                    false, false);
    }

#pragma unroll
    for (int i = 0; i < 2; ++i)
#pragma unroll
        for (int j = 0; j < 2; ++j) {
            int gn = n0 + wn + j * 16 + lr;
            float bv = bias ? bias[gn] : 0.f;
#pragma unroll
            for (int r = 0; r < 8; ++r) {
                int gm = m0 + wm + i * 16 + r + hi * 8;
                float v = acc[i][j][r] + bv;
                if (leaky && v < 0.f) v *= 0.01f;
                Cf[(size_t)gm * ldc + gn] = v;
                if (Cb) Cb[(size_t)gm * ldc + gn] = f2bf(v);
            }
        }
}

// ---------------------------------------------------------------------------
// fp32 WMMA GEMM (16x16x4), bounds checked. One 16x16 C tile per wave.
// ---------------------------------------------------------------------------
__global__ __launch_bounds__(128)
void k_gemm_f32(const float* __restrict__ A, int lda,
                const float* __restrict__ B, int ldb,
                float* __restrict__ C, int ldc, int M, int N, int K) {
    int lane = threadIdx.x & 31, wid = threadIdx.x >> 5;
    int tilesN = (N + 15) >> 4;
    int tilesM = (M + 15) >> 4;
    int tile = blockIdx.x * 4 + wid;
    if (tile >= tilesM * tilesN) return;   // wave-uniform exit
    int tm = tile / tilesN, tn = tile % tilesN;
    int hi = (lane >> 4) & 1;
    int lr = lane & 15;
    int m = tm * 16 + lr;
    int n = tn * 16 + lr;
    v8f acc = {};
    for (int k0 = 0; k0 < K; k0 += 4) {
        int ka = k0 + hi * 2;
        v2f a, b;
        a.x = (m < M && ka     < K) ? A[(size_t)m * lda + ka]       : 0.f;
        a.y = (m < M && ka + 1 < K) ? A[(size_t)m * lda + ka + 1]   : 0.f;
        b.x = (n < N && ka     < K) ? B[(size_t)ka * ldb + n]       : 0.f;
        b.y = (n < N && ka + 1 < K) ? B[(size_t)(ka + 1) * ldb + n] : 0.f;
        acc = __builtin_amdgcn_wmma_f32_16x16x4_f32(
            false, a, false, b, (short)0, acc, false, false);
    }
#pragma unroll
    for (int r = 0; r < 8; ++r) {
        int gm = tm * 16 + r + hi * 8;
        if (gm < M && n < N) C[(size_t)gm * ldc + n] = acc[r];
    }
}

// ---------------------------------------------------------------------------
// Cyclic Jacobi eigensolver on symmetric 256x256 G. G lives in 256KB of LDS
// (fits the 320KB CDNA5 WGP LDS); eigenvectors V accumulate in global (L2).
// ---------------------------------------------------------------------------
__global__ __launch_bounds__(256)
void k_jacobi(const float* __restrict__ Gin, float* __restrict__ V,
              float* __restrict__ lam, int nsweeps) {
    extern __shared__ float sG[];              // 256*256 floats = 256 KB
    __shared__ float sc[128], ss[128];
    __shared__ int   sp[128], sq[128];
    const int tid = threadIdx.x;
    for (int i = tid; i < 256 * 256; i += 256) {
        sG[i] = Gin[i];
        V[i]  = ((i >> 8) == (i & 255)) ? 1.f : 0.f;
    }
    __syncthreads();
    for (int sw = 0; sw < nsweeps; ++sw)
        for (int r = 0; r < 255; ++r) {
            if (tid < 128) {
                int k = tid, a, b;
                if (k == 0) { a = 255; b = r; }
                else { a = (r + k) % 255; b = (r - k + 255) % 255; }
                int p = a < b ? a : b, q = a < b ? b : a;
                sp[k] = p; sq[k] = q;
                float app = sG[p * 256 + p], aqq = sG[q * 256 + q];
                float apq = sG[p * 256 + q];
                float c = 1.f, s = 0.f;
                if (fabsf(apq) > 1e-12f) {
                    float tau = (aqq - app) / (2.f * apq);
                    float t = (tau >= 0.f ? 1.f : -1.f) /
                              (fabsf(tau) + sqrtf(1.f + tau * tau));
                    c = rsqrtf(1.f + t * t);
                    s = t * c;
                }
                sc[k] = c; ss[k] = s;
            }
            __syncthreads();
            {   // row update: G <- J^T G (disjoint rows per pair)
                int pr = tid >> 1, half = tid & 1;
                int p = sp[pr], q = sq[pr];
                float c = sc[pr], s = ss[pr];
                for (int col = half * 128; col < half * 128 + 128; ++col) {
                    float gp = sG[p * 256 + col], gq = sG[q * 256 + col];
                    sG[p * 256 + col] = c * gp - s * gq;
                    sG[q * 256 + col] = s * gp + c * gq;
                }
            }
            __syncthreads();
            {   // col update: G <- G J, and V <- V J
                int pr = tid >> 1, half = tid & 1;
                int p = sp[pr], q = sq[pr];
                float c = sc[pr], s = ss[pr];
                for (int row = half * 128; row < half * 128 + 128; ++row) {
                    float gp = sG[row * 256 + p], gq = sG[row * 256 + q];
                    sG[row * 256 + p] = c * gp - s * gq;
                    sG[row * 256 + q] = s * gp + c * gq;
                    float vp = V[row * 256 + p], vq = V[row * 256 + q];
                    V[row * 256 + p] = c * vp - s * vq;
                    V[row * 256 + q] = s * vp + c * vq;
                }
            }
            __syncthreads();
        }
    lam[tid] = sG[tid * 256 + tid];
}

// ---------------------------------------------------------------------------
// Sort eigenpairs desc by s=sqrt(lam), low-rank mask, gather U variants.
// ---------------------------------------------------------------------------
__global__ __launch_bounds__(256)
void k_eigsort(const float* __restrict__ lam, const float* __restrict__ V,
               float* __restrict__ Usort, float* __restrict__ UsortT,
               float* __restrict__ Umm, float* __restrict__ sv,
               float* __restrict__ maskv, float* __restrict__ minv2,
               float* __restrict__ Uout) {
    __shared__ float key[256];
    __shared__ int   kidx[256];
    __shared__ float scan[256];
    __shared__ int   rShared;
    int tid = threadIdx.x;
    key[tid]  = sqrtf(fmaxf(lam[tid], 0.f));
    kidx[tid] = tid;
    __syncthreads();
    for (int sz = 2; sz <= 256; sz <<= 1)
        for (int st = sz >> 1; st > 0; st >>= 1) {
            int ix = tid ^ st;
            if (ix > tid) {
                bool desc = ((tid & sz) == 0);
                float k0 = key[tid], k1 = key[ix];
                if (desc ? (k0 < k1) : (k0 > k1)) {
                    key[tid] = k1; key[ix] = k0;
                    int t = kidx[tid]; kidx[tid] = kidx[ix]; kidx[ix] = t;
                }
            }
            __syncthreads();
        }
    scan[tid] = key[tid];
    __syncthreads();
    for (int off = 1; off < 256; off <<= 1) {
        float add = (tid >= off) ? scan[tid - off] : 0.f;
        __syncthreads();
        scan[tid] += add;
        __syncthreads();
    }
    float total = scan[255];
    if (tid == 0) rShared = 256;
    __syncthreads();
    if (scan[tid] >= 0.999f * total) atomicMin(&rShared, tid + 1);
    __syncthreads();
    int   rr = rShared;
    float s  = key[tid];
    float mk = (tid < rr) ? 1.f : 0.f;
    sv[tid]    = s;
    maskv[tid] = mk;
    minv2[tid] = (s > 0.f) ? mk / (s * s) : 0.f;
    int src = kidx[tid];
    for (int i = 0; i < 256; ++i) {
        float u = V[i * 256 + src];
        Usort[i * 256 + tid]  = u;
        UsortT[tid * 256 + i] = u;
        Umm[i * 256 + tid]    = u * mk;
        Uout[i * 256 + tid]   = u;
    }
}

// z0 = diag(mask) U^T Z[0,:]  -> written as column 0 of P (256 x 8192)
__global__ __launch_bounds__(256)
void k_z0(const float* __restrict__ Usort, const float* __restrict__ maskv,
          const float* __restrict__ Zrow0, float* __restrict__ P) {
    int i = threadIdx.x;
    float acc = 0.f;
    for (int j = 0; j < 256; ++j) acc += Usort[j * 256 + i] * Zrow0[j];
    P[(size_t)i * 8192] = acc * maskv[i];
}

// Atilde[i][j] = mask_i * T2[i][j] * (mask_j / s_j^2)
__global__ __launch_bounds__(256)
void k_scaleA(const float* __restrict__ T2, const float* __restrict__ maskv,
              const float* __restrict__ minv2, float* __restrict__ Aout,
              float* __restrict__ Aws) {
    int j = threadIdx.x, i = blockIdx.x;
    float v = maskv[i] * T2[i * 256 + j] * minv2[j];
    Aout[i * 256 + j] = v;
    Aws[i * 256 + j]  = v;
}

// ---------------------------------------------------------------------------
// Host-side orchestration
// ---------------------------------------------------------------------------
extern "C" void kernel_launch(void* const* d_in, const int* in_sizes, int n_in,
                              void* d_out, int out_size, void* d_ws, size_t ws_size,
                              hipStream_t stream) {
    (void)in_sizes; (void)n_in; (void)out_size; (void)ws_size;
    const int ORIG = 4096, LAT = 256, T = 8192, HALF = 4096;

    const float* X    = (const float*)d_in[1];
    const float* eW[4] = { (const float*)d_in[2], (const float*)d_in[4],
                           (const float*)d_in[6], (const float*)d_in[8] };
    const float* eB[4] = { (const float*)d_in[3], (const float*)d_in[5],
                           (const float*)d_in[7], (const float*)d_in[9] };
    const float* dW[4] = { (const float*)d_in[10], (const float*)d_in[12],
                           (const float*)d_in[14], (const float*)d_in[16] };
    const float* dB[4] = { (const float*)d_in[11], (const float*)d_in[13],
                           (const float*)d_in[15], (const float*)d_in[17] };

    float* out      = (float*)d_out;
    float* outXpred = out;
    float* outAt    = out + (size_t)T * ORIG;
    float* outU     = outAt + (size_t)LAT * LAT;

    char* ws = (char*)d_ws;
    size_t off = 0;
    auto alloc = [&](size_t bytes) -> char* {
        char* p = ws + off;
        off = (off + bytes + 255) & ~(size_t)255;
        return p;
    };
    uint16_t* Xbf   = (uint16_t*)alloc((size_t)T * ORIG * 2);
    uint16_t* W1bf  = (uint16_t*)alloc((size_t)ORIG * LAT * 2);
    uint16_t* W2bf  = (uint16_t*)alloc((size_t)LAT * LAT * 2);
    uint16_t* W3bf  = (uint16_t*)alloc((size_t)LAT * LAT * 2);
    uint16_t* W4bf  = (uint16_t*)alloc((size_t)LAT * LAT * 2);
    uint16_t* DW1bf = (uint16_t*)alloc((size_t)LAT * LAT * 2);
    uint16_t* DW2bf = (uint16_t*)alloc((size_t)LAT * LAT * 2);
    uint16_t* DW3bf = (uint16_t*)alloc((size_t)LAT * LAT * 2);
    uint16_t* DW4bf = (uint16_t*)alloc((size_t)LAT * ORIG * 2);
    float*    Hf32  = (float*)   alloc((size_t)T * LAT * 4);
    uint16_t* Habf  = (uint16_t*)alloc((size_t)T * LAT * 2);
    uint16_t* Hbbf  = (uint16_t*)alloc((size_t)T * LAT * 2);
    float*    Zf32  = (float*)   alloc((size_t)T * LAT * 4);
    uint16_t* Zbf   = (uint16_t*)alloc((size_t)T * LAT * 2);
    uint16_t* Ztbf  = (uint16_t*)alloc((size_t)LAT * T * 2);
    float*    G     = (float*)   alloc((size_t)LAT * LAT * 4);
    float*    Mx    = (float*)   alloc((size_t)LAT * LAT * 4);
    float*    Vg    = (float*)   alloc((size_t)LAT * LAT * 4);
    float*    Usort = (float*)   alloc((size_t)LAT * LAT * 4);
    float*    UsortT= (float*)   alloc((size_t)LAT * LAT * 4);
    float*    Umm   = (float*)   alloc((size_t)LAT * LAT * 4);
    float*    T1    = (float*)   alloc((size_t)LAT * LAT * 4);
    float*    T2b   = (float*)   alloc((size_t)LAT * LAT * 4);
    float*    Aws   = (float*)   alloc((size_t)LAT * LAT * 4);
    float*    ApA   = (float*)   alloc((size_t)LAT * LAT * 4);
    float*    ApB   = (float*)   alloc((size_t)LAT * LAT * 4);
    float*    lam   = (float*)   alloc(1024);
    float*    sv    = (float*)   alloc(1024);
    float*    maskv = (float*)   alloc(1024);
    float*    minv2 = (float*)   alloc(1024);
    float*    P     = (float*)   alloc((size_t)LAT * T * 4);
    float*    S     = (float*)   alloc((size_t)LAT * T * 4);
    uint16_t* Xlatbf= (uint16_t*)alloc((size_t)T * LAT * 2);

    auto cvt = [&](const float* src, uint16_t* dst, long n) {
        k_f2bf<<<dim3((unsigned)((n + 255) / 256)), 256, 0, stream>>>(src, dst, n);
    };
    auto gemmB = [&](const uint16_t* A, int lda, const uint16_t* B, int ldb,
                     const float* bias, float* Cf, uint16_t* Cb, int ldc,
                     int M, int N, int K, int leaky) {
        k_gemm_bf16<<<dim3(N / 64, M / 64), 128, 0, stream>>>(
            A, lda, B, ldb, bias, Cf, Cb, ldc, M, N, K, leaky);
    };
    auto gemmF = [&](const float* A, int lda, const float* B, int ldb,
                     float* C, int ldc, int M, int N, int K) {
        int tiles = ((M + 15) / 16) * ((N + 15) / 16);
        k_gemm_f32<<<dim3((tiles + 3) / 4), 128, 0, stream>>>(
            A, lda, B, ldb, C, ldc, M, N, K);
    };
    auto trans = [&](const float* in, uint16_t* o, int R, int C) {
        k_transpose_f2bf<<<dim3(C / 32, R / 32), 256, 0, stream>>>(in, o, R, C);
    };

    // 1) convert inputs to bf16
    cvt(X, Xbf, (long)T * ORIG);
    cvt(eW[0], W1bf, (long)ORIG * LAT);
    cvt(eW[1], W2bf, (long)LAT * LAT);
    cvt(eW[2], W3bf, (long)LAT * LAT);
    cvt(eW[3], W4bf, (long)LAT * LAT);
    cvt(dW[0], DW1bf, (long)LAT * LAT);
    cvt(dW[1], DW2bf, (long)LAT * LAT);
    cvt(dW[2], DW3bf, (long)LAT * LAT);
    cvt(dW[3], DW4bf, (long)LAT * ORIG);

    // 2) encoder MLP (bf16 WMMA, fp32 accumulate)
    gemmB(Xbf, ORIG, W1bf, LAT, eB[0], Hf32, Habf, LAT, T, LAT, ORIG, 1);
    gemmB(Habf, LAT, W2bf, LAT, eB[1], Hf32, Hbbf, LAT, T, LAT, LAT, 1);
    gemmB(Hbbf, LAT, W3bf, LAT, eB[2], Hf32, Habf, LAT, T, LAT, LAT, 1);
    gemmB(Habf, LAT, W4bf, LAT, eB[3], Zf32, Zbf, LAT, T, LAT, LAT, 0);

    // 3) Zt (bf16), Gram matrices G = X1 X1^T, M = X2 X1^T  (K = 4096)
    trans(Zf32, Ztbf, T, LAT);
    gemmB(Ztbf, T, Zbf, LAT, nullptr, G, nullptr, LAT, LAT, LAT, HALF, 0);
    gemmB(Ztbf + HALF, T, Zbf, LAT, nullptr, Mx, nullptr, LAT, LAT, LAT, HALF, 0);

    // 4) Jacobi eigensolver: G in 256KB LDS (fits 320KB WGP LDS)
    k_jacobi<<<1, 256, 256 * 256 * sizeof(float), stream>>>(G, Vg, lam, 8);
    k_eigsort<<<1, 256, 0, stream>>>(lam, Vg, Usort, UsortT, Umm,
                                     sv, maskv, minv2, outU);

    // 5) Atilde = diag(mask) U^T M U diag(mask/s^2)   (fp32 WMMA)
    gemmF(UsortT, LAT, Mx, LAT, T1, LAT, LAT, LAT, LAT);
    gemmF(T1, LAT, Usort, LAT, T2b, LAT, LAT, LAT, LAT);
    k_scaleA<<<256, 256, 0, stream>>>(T2b, maskv, minv2, outAt, Aws);

    // 6) Krylov doubling: P[:, w:2w] = A^w P[:, :w];  A^w by squaring
    k_z0<<<1, 256, 0, stream>>>(Usort, maskv, Zf32, P);
    {
        const float* cur = Aws;
        float* bufs[2] = { ApA, ApB };
        int bsel = 0;
        for (int w = 1; w < T; w <<= 1) {
            int wn = (w <= T - w) ? w : (T - w);
            gemmF(cur, LAT, P, T, P + w, T, LAT, wn, LAT);
            if (w * 2 < T) {
                float* nxt = bufs[bsel]; bsel ^= 1;
                gemmF(cur, LAT, cur, LAT, nxt, LAT, LAT, LAT, LAT);
                cur = nxt;
            }
        }
    }

    // 7) X_pred_lat^T = Umm @ P, then transpose to (T x LAT) bf16
    gemmF(Umm, LAT, P, T, S, T, LAT, T, LAT);
    trans(S, Xlatbf, LAT, T);

    // 8) decoder MLP; final layer writes fp32 straight into d_out
    gemmB(Xlatbf, LAT, DW1bf, LAT, dB[0], Hf32, Habf, LAT, T, LAT, LAT, 1);
    gemmB(Habf, LAT, DW2bf, LAT, dB[1], Hf32, Hbbf, LAT, T, LAT, LAT, 1);
    gemmB(Hbbf, LAT, DW3bf, LAT, dB[2], Hf32, Habf, LAT, T, LAT, LAT, 1);
    gemmB(Habf, LAT, DW4bf, ORIG, dB[3], outXpred, nullptr, ORIG,
          T, ORIG, LAT, 0);
}
